// encoder_23184233464477
// MI455X (gfx1250) — matrix operand order repaired and checked by reading, not compiled, not run on hardware
//
#include <hip/hip_runtime.h>
#include <hip/hip_bf16.h>
#include <math.h>

// ---------------------------------------------------------------------------
// CDNA5 (gfx1250) implementation of the SO(3)-steerable 3D CNN reference.
// Convolutions run as implicit-GEMM on v_wmma_f32_16x16x32_f16 (wave32),
// 4 N-tiles per wave (A-fragment reuse), weights staged to LDS via
// global_load_async_to_lds_b128, vectorized LDS fragment packing,
// branch-free clamped im2col staging with 32-bit offsets.
// ---------------------------------------------------------------------------

typedef __attribute__((ext_vector_type(16))) _Float16 v16h;
typedef __attribute__((ext_vector_type(8)))  _Float16 v8h;
typedef __attribute__((ext_vector_type(8)))  float    v8f;
typedef __attribute__((ext_vector_type(4)))  int      v4i;

#define CONV_TM 128   // cout per workgroup (8 waves x 16)
#define CONV_TN 64    // spatial positions per workgroup (4 x 16 per wave)
#define CONV_TK 32    // K-step (matches 16x16x32 f16 WMMA)

#if __has_builtin(__builtin_amdgcn_global_load_async_to_lds_b128)
#define HAVE_ASYNC_LDS 1
#else
#define HAVE_ASYNC_LDS 0
#endif

__device__ __forceinline__ void wait_asynccnt0() {
#if __has_builtin(__builtin_amdgcn_s_wait_asynccnt)
    __builtin_amdgcn_s_wait_asynccnt(0);
#else
    asm volatile("s_wait_asynccnt 0" ::: "memory");
#endif
}

// ---------------------------------------------------------------------------
// Implicit-GEMM conv3d:  Y[co, p] (+)= sum_k W[co, k] * patch(X)[k, p]
// k = ci*KS^3 + kd*KS^2 + kh*KS + kw ; p = linear output voxel index.
// f32 in/out, f16 multiply, f32 accumulate via WMMA.
// All tensors are < 2^31 elements, so 32-bit offsets are used throughout.
// ---------------------------------------------------------------------------
template <int KS>
__global__ void __launch_bounds__(256)
conv3d_wmma(const float* __restrict__ X, const float* __restrict__ W,
            float* __restrict__ Y,
            int Cin, int Cout, int stride, int pad,
            int Din, int Hin, int Win, int Do, int Ho, int Wo,
            int accumulate)
{
    constexpr int KV = KS * KS * KS;

    // row 128 of sAf is a dedicated always-zero row (for cout-overflow lanes)
    __shared__ __align__(128) float    sAf[CONV_TM + 1][CONV_TK]; // ~16.5 KB
    __shared__ __align__(32)  _Float16 sBt[CONV_TN][CONV_TK];     //    4 KB

    const int tid  = threadIdx.x;
    const int lane = tid & 31;
    const int wv   = tid >> 5;          // wave id 0..7
    const int n    = lane & 15;         // output column within 16-tile
    const int hf   = lane >> 4;         // lane half (0/1)

    const int K    = Cin * KV;
    const int P    = Do * Ho * Wo;
    const int HWin = Hin * Win;
    const int HWo  = Ho * Wo;

    const int p0 = blockIdx.x * CONV_TN;
    const int c0 = blockIdx.y * CONV_TM;

    // ---- per-thread B-staging assignment (fixed across K loop) ------------
    const int nnB  = tid & 63;          // position column this thread stages
    const int klB0 = (tid >> 6) * 8;    // 8 consecutive k-rows per thread
    const int pB   = p0 + nnB;
    const bool pBok = (pB < P);
    int idB0 = 0, ihB0 = 0, iwB0 = 0;   // input coords at (kd,kh,kw)=(0,0,0)
    if (pBok) {
        idB0 = (pB / HWo) * stride - pad;
        ihB0 = ((pB / Wo) % Ho) * stride - pad;
        iwB0 = (pB % Wo) * stride - pad;
    }

    // ---- per-thread A-staging assignment ----------------------------------
    const int rowA  = tid >> 1;
    const int ksegA = (tid & 1) * 16;
    const int coA   = c0 + rowA;
    const bool coAok = (coA < Cout);
    const int coAc   = coAok ? coA : (Cout - 1);        // clamped row
    const long maxoffA = (((long)Cout * K) - 4) & ~3L;  // 16B-aligned clamp
    const bool asyncOK = ((K & 3) == 0);

    // zero the spare A row once (visible after first loop-top barrier)
    if (tid < CONV_TK) sAf[CONV_TM][tid] = 0.f;

    // ---- accumulators: 4 N-tiles of 16x16 ---------------------------------
    v8f acc[4];
#pragma unroll
    for (int nt = 0; nt < 4; ++nt)
        acc[nt] = (v8f){0.f, 0.f, 0.f, 0.f, 0.f, 0.f, 0.f, 0.f};
    if (accumulate) {
#pragma unroll
        for (int nt = 0; nt < 4; ++nt) {
#pragma unroll
            for (int r = 0; r < 8; ++r) {
                const int co = c0 + wv * 16 + 8 * hf + r;
                const int p  = p0 + nt * 16 + n;
                acc[nt][r] = (co < Cout && p < P) ? Y[co * P + p] : 0.f;
            }
        }
    }

    const int mrow    = wv * 16 + (lane & 15);
    const int mrowEff = ((c0 + mrow) < Cout) ? mrow : CONV_TM;  // zero row

    const int ksteps = (K + CONV_TK - 1) / CONV_TK;
    for (int kk = 0; kk < ksteps; ++kk) {
        const int k0 = kk * CONV_TK;
        __syncthreads();    // previous-iteration readers are done

        // ---- stage B tile: im2col patches (64 positions x 32 k), f16 ------
        // Branch-free: 32-bit index select (v_cndmask), clamped to element 0.
        {
            float fv[8];
            bool  ok[8];
#pragma unroll
            for (int j = 0; j < 8; ++j) {
                const int kl = klB0 + j;
                const int k  = k0 + kl;
                const int ci = k / KV;
                const int r  = k - ci * KV;
                const int kd = r / (KS * KS);
                const int kh = (r / KS) % KS;
                const int kw = r % KS;
                const int id = idB0 + kd;
                const int ih = ihB0 + kh;
                const int iw = iwB0 + kw;
                const bool o = (k < K) & pBok &
                               ((unsigned)id < (unsigned)Din) &
                               ((unsigned)ih < (unsigned)Hin) &
                               ((unsigned)iw < (unsigned)Win);
                int idx32 = ci * (Din * HWin) + id * HWin + ih * Win + iw;
                idx32 = o ? idx32 : 0;          // single v_cndmask_b32
                ok[j] = o;
                fv[j] = X[idx32];               // unconditional clamped load
            }
            v8h tmp;
#pragma unroll
            for (int j = 0; j < 8; ++j)
                tmp[j] = (_Float16)(ok[j] ? fv[j] : 0.f);
            *(v8h*)&sBt[nnB][klB0] = tmp;       // one ds_store_b128
        }

        // ---- stage A tile: weights (128 cout x 32 k), f32 -----------------
#if HAVE_ASYNC_LDS
        if (asyncOK) {
            const long base = (long)coAc * K + k0 + ksegA;
#pragma unroll
            for (int jj = 0; jj < 4; ++jj) {
                long off = base + 4 * jj;           // 16B-aligned (K % 4 == 0)
                if (off > maxoffA) off = maxoffA;   // clamp into weight buffer
                __builtin_amdgcn_global_load_async_to_lds_b128(
                    (v4i*)(W + off), (v4i*)&sAf[rowA][ksegA + 4 * jj], 0, 0);
            }
            wait_asynccnt0();
        } else
#endif
        {
            const long base = (long)coAc * K;
#pragma unroll
            for (int j = 0; j < 16; ++j) {
                const int k = k0 + ksegA + j;
                sAf[rowA][ksegA + j] = (coAok && k < K) ? W[base + k] : 0.f;
            }
        }
        // K-tail: zero-patch LDS so the pack loop needs no guards (uniform br)
        if (K - k0 < CONV_TK) {
#pragma unroll
            for (int j = 0; j < 16; ++j)
                if (k0 + ksegA + j >= K) sAf[rowA][ksegA + j] = 0.f;
        }
        __syncthreads();

        // ---- pack fragments in contiguous k order (k = 16*hf + s), A and B alike ------------
        // A: two contiguous 8-float runs per lane -> 2 vector LDS loads
        const v8f a0 = *(const v8f*)&sAf[mrowEff][16 * hf];
        const v8f a1 = *(const v8f*)&sAf[mrowEff][16 * hf + 8];
        v16h afrag;
#pragma unroll
        for (int j = 0; j < 8; ++j) {
            afrag[j]     = (_Float16)a0[j];
            afrag[8 + j] = (_Float16)a1[j];
        }
#pragma unroll
        for (int nt = 0; nt < 4; ++nt) {
            // B: 16 contiguous f16 per lane -> one 32B LDS load
            const v16h bfrag = *(const v16h*)&sBt[nt * 16 + n][16 * hf];
            acc[nt] = __builtin_amdgcn_wmma_f32_16x16x32_f16(
                          false, afrag, false, bfrag, (short)0, acc[nt],
                          false, false);
        }
    }

    // ---- store C fragments -------------------------------------------------
#pragma unroll
    for (int nt = 0; nt < 4; ++nt) {
#pragma unroll
        for (int r = 0; r < 8; ++r) {
            const int co = c0 + wv * 16 + 8 * hf + r;
            const int p  = p0 + nt * 16 + n;
            if (co < Cout && p < P) Y[co * P + p] = acc[nt][r];
        }
    }
}

// ---------------------------------------------------------------------------
// IID BatchNorm: per-field stats (mean over trivial coord, analytic var).
// X layout: [nf*35][P].  stats[2f] = mu, stats[2f+1] = gamma/sqrt(var+eps)
// ---------------------------------------------------------------------------
__global__ void __launch_bounds__(256)
bn_stats_kernel(const float* __restrict__ X, const float* __restrict__ gamma,
                float* __restrict__ stats, int P)
{
    __shared__ float s0[256], s1[256];
    const int f = blockIdx.x;
    const float* base = X + (long)f * 35 * P;
    float sum0 = 0.f, sumsq = 0.f;
    const int N = 35 * P;
    for (int i = threadIdx.x; i < N; i += 256) {
        float v = base[i];
        sumsq += v * v;
        if (i < P) sum0 += v;            // d == 0 rows occupy the first P entries
    }
    s0[threadIdx.x] = sum0;
    s1[threadIdx.x] = sumsq;
    __syncthreads();
    for (int off = 128; off > 0; off >>= 1) {
        if (threadIdx.x < off) {
            s0[threadIdx.x] += s0[threadIdx.x + off];
            s1[threadIdx.x] += s1[threadIdx.x + off];
        }
        __syncthreads();
    }
    if (threadIdx.x == 0) {
        const float mu  = s0[0] / (float)P;
        const float var = (s1[0] - (float)P * mu * mu) / (35.f * (float)P);
        stats[2 * f]     = mu;
        stats[2 * f + 1] = gamma[f] * rsqrtf(var + 1e-5f);
    }
}

__global__ void __launch_bounds__(256)
bn_apply_kernel(float* __restrict__ X, const float* __restrict__ stats,
                int nf, int P)
{
    const int idx = blockIdx.x * 256 + threadIdx.x;
    const int N = nf * 35 * P;
    if (idx >= N) return;
    const int f   = idx / (35 * P);
    const int rem = idx - f * 35 * P;
    const int d   = rem / P;
    const float mu  = stats[2 * f];
    const float inv = stats[2 * f + 1];
    float v = X[idx];
    v = (d == 0 ? v - mu : v) * inv;
    X[idx] = v;
}

// ---------------------------------------------------------------------------
// Fourier ELU: per (field, voxel):  y = Ainv @ elu(A @ x), in place.
// A: [96][35], Ainv: [35][96], X layout [nf*35][P].
// ---------------------------------------------------------------------------
__global__ void __launch_bounds__(128)
fourier_elu_kernel(float* __restrict__ X, const float* __restrict__ A,
                   const float* __restrict__ Ainv, int nf, int P)
{
    __shared__ float sA[96 * 35];
    __shared__ float sAi[35 * 96];
    for (int i = threadIdx.x; i < 96 * 35; i += 128) {
        sA[i]  = A[i];
        sAi[i] = Ainv[i];
    }
    __syncthreads();

    const int idx = blockIdx.x * 128 + threadIdx.x;
    if (idx >= nf * P) return;
    const int f = idx / P;
    const int p = idx - f * P;
    float* base = X + (long)f * 35 * P + p;

    float xv[35], yv[35];
#pragma unroll
    for (int d = 0; d < 35; ++d) { xv[d] = base[(long)d * P]; yv[d] = 0.f; }

    for (int s = 0; s < 96; ++s) {
        float g = 0.f;
#pragma unroll
        for (int d = 0; d < 35; ++d) g += sA[s * 35 + d] * xv[d];
        g = g > 0.f ? g : expm1f(g);
#pragma unroll
        for (int d = 0; d < 35; ++d) yv[d] += sAi[d * 96 + s] * g;
    }
#pragma unroll
    for (int d = 0; d < 35; ++d) base[(long)d * P] = yv[d];
}

// ---------------------------------------------------------------------------
// Depthwise gaussian blur (sigma=0.33, k3, stride2, pad1).
// ---------------------------------------------------------------------------
__global__ void __launch_bounds__(256)
gauss_blur_kernel(const float* __restrict__ X, float* __restrict__ Y,
                  int C, int Din, int Hin, int Win, int Do, int Ho, int Wo)
{
    const int idx = blockIdx.x * 256 + threadIdx.x;
    const int P = Do * Ho * Wo;
    if (idx >= C * P) return;
    const int c = idx / P;
    const int p = idx - c * P;
    const int od = p / (Ho * Wo);
    const int oh = (p / Wo) % Ho;
    const int ow = p % Wo;

    const float g1 = expf(-1.f / (2.f * 0.33f * 0.33f));
    const float s1 = 1.f + 2.f * g1;
    const float norm = 1.f / (s1 * s1 * s1);

    float acc = 0.f;
    const long cbase = (long)c * Din * Hin * Win;
#pragma unroll
    for (int kd = 0; kd < 3; ++kd) {
        const int id = od * 2 - 1 + kd;
        if ((unsigned)id >= (unsigned)Din) continue;
        const float wd = (kd == 1) ? 1.f : g1;
#pragma unroll
        for (int kh = 0; kh < 3; ++kh) {
            const int ih = oh * 2 - 1 + kh;
            if ((unsigned)ih >= (unsigned)Hin) continue;
            const float wh = (kh == 1) ? 1.f : g1;
#pragma unroll
            for (int kw = 0; kw < 3; ++kw) {
                const int iw = ow * 2 - 1 + kw;
                if ((unsigned)iw >= (unsigned)Win) continue;
                const float ww = (kw == 1) ? 1.f : g1;
                acc += wd * wh * ww * X[cbase + ((long)id * Hin + ih) * Win + iw];
            }
        }
    }
    Y[idx] = acc * norm;
}

__global__ void __launch_bounds__(256)
add_kernel(float* __restrict__ Y, const float* __restrict__ S, int N)
{
    const int idx = blockIdx.x * 256 + threadIdx.x;
    if (idx < N) Y[idx] += S[idx];
}

// ---------------------------------------------------------------------------
// Quotient-Fourier pooling head: out[c,p] = sum_s Aout[s]*elu(Aq[s,:]@x[c,:,p])
// X layout [32*9][P], P = 8.
// ---------------------------------------------------------------------------
__global__ void __launch_bounds__(256)
quotient_pool_kernel(const float* __restrict__ X, const float* __restrict__ Aq,
                     const float* __restrict__ Aout, float* __restrict__ out,
                     int P)
{
    const int idx = blockIdx.x * 256 + threadIdx.x;
    if (idx >= 32 * P) return;
    const int c = idx / P;
    const int p = idx - c * P;
    float xv[9];
#pragma unroll
    for (int d = 0; d < 9; ++d) xv[d] = X[(c * 9 + d) * P + p];
    float o = 0.f;
    for (int s = 0; s < 24; ++s) {
        float g = 0.f;
#pragma unroll
        for (int d = 0; d < 9; ++d) g += Aq[s * 9 + d] * xv[d];
        g = g > 0.f ? g : expm1f(g);
        o += Aout[s] * g;
    }
    out[idx] = o;
}

// ---------------------------------------------------------------------------
// Host orchestration
// ---------------------------------------------------------------------------
struct BlockCfg { int ci, cr, co, s, nf, szi, szo; };
static const BlockCfg g_blocks[6] = {
    {  52,  70,  320, 1,  2, 32, 32},
    { 320,  70,  320, 2,  2, 32, 16},
    { 320, 175,  640, 1,  5, 16, 16},
    { 640, 350,  640, 2, 10, 16,  8},
    { 640, 350, 1280, 1, 10,  8,  8},
    {1280, 700, 1280, 2, 20,  8,  4},
};

extern "C" void kernel_launch(void* const* d_in, const int* in_sizes, int n_in,
                              void* d_out, int out_size, void* d_ws, size_t ws_size,
                              hipStream_t stream)
{
    (void)in_sizes; (void)n_in; (void)out_size; (void)ws_size;

    const float* x   = (const float*)d_in[0];
    const float* w0  = (const float*)d_in[1];
    const float* wc1[6]  = {(const float*)d_in[2],  (const float*)d_in[6],
                            (const float*)d_in[9],  (const float*)d_in[13],
                            (const float*)d_in[16], (const float*)d_in[20]};
    const float* gam[6]  = {(const float*)d_in[3],  (const float*)d_in[7],
                            (const float*)d_in[10], (const float*)d_in[14],
                            (const float*)d_in[17], (const float*)d_in[21]};
    const float* wc2[6]  = {(const float*)d_in[4],  (const float*)d_in[8],
                            (const float*)d_in[11], (const float*)d_in[15],
                            (const float*)d_in[18], (const float*)d_in[22]};
    const float* wsk[6]  = {(const float*)d_in[5],  nullptr,
                            (const float*)d_in[12], nullptr,
                            (const float*)d_in[19], nullptr};
    const float* A    = (const float*)d_in[23];
    const float* Ainv = (const float*)d_in[24];
    const float* wf   = (const float*)d_in[25];
    const float* Aq   = (const float*)d_in[26];
    const float* Aout = (const float*)d_in[27];

    char* ws = (char*)d_ws;
    const size_t MB = 1024ull * 1024ull;
    float* bufX  = (float*)(ws + 0);         // activations (max 40 MB)
    float* bufY  = (float*)(ws + 42 * MB);   // conv2 / next activations
    float* bufR  = (float*)(ws + 84 * MB);   // residual intermediate (max 8.75 MB)
    float* bufS  = (float*)(ws + 94 * MB);   // blurred skip (max 5 MB)
    float* stats = (float*)(ws + 100 * MB);  // bn stats (<=40 floats)

    auto conv = [&](const float* Xp, const float* Wp, float* Yp,
                    int Cin, int Cout, int ks, int stride, int pad,
                    int szi, int szo, int accumulate) {
        const int P = szo * szo * szo;
        dim3 grid((P + CONV_TN - 1) / CONV_TN, (Cout + CONV_TM - 1) / CONV_TM);
        if (ks == 1)
            conv3d_wmma<1><<<grid, 256, 0, stream>>>(
                Xp, Wp, Yp, Cin, Cout, stride, pad,
                szi, szi, szi, szo, szo, szo, accumulate);
        else if (ks == 3)
            conv3d_wmma<3><<<grid, 256, 0, stream>>>(
                Xp, Wp, Yp, Cin, Cout, stride, pad,
                szi, szi, szi, szo, szo, szo, accumulate);
        else
            conv3d_wmma<5><<<grid, 256, 0, stream>>>(
                Xp, Wp, Yp, Cin, Cout, stride, pad,
                szi, szi, szi, szo, szo, szo, accumulate);
    };

    // ---- stem: k5 p2, 1 -> 52 ch at 32^3 ----------------------------------
    conv(x, w0, bufX, 1, 52, 5, 1, 2, 32, 32, 0);

    float* cur = bufX;
    float* nxt = bufY;

    for (int i = 0; i < 6; ++i) {
        const BlockCfg b = g_blocks[i];
        const int Pi = b.szi * b.szi * b.szi;
        const int Po = b.szo * b.szo * b.szo;

        // residual branch: conv1 -> IID BN -> Fourier ELU -> conv2
        conv(cur, wc1[i], bufR, b.ci, b.cr, 3, 1, 1, b.szi, b.szi, 0);

        bn_stats_kernel<<<b.nf, 256, 0, stream>>>(bufR, gam[i], stats, Pi);
        {
            const int N = b.nf * 35 * Pi;
            bn_apply_kernel<<<(N + 255) / 256, 256, 0, stream>>>(bufR, stats, b.nf, Pi);
        }
        {
            const int N = b.nf * Pi;
            fourier_elu_kernel<<<(N + 127) / 128, 128, 0, stream>>>(bufR, A, Ainv, b.nf, Pi);
        }
        conv(bufR, wc2[i], nxt, b.cr, b.co, 3, b.s, 1, b.szi, b.szo, 0);

        // skip branch
        if (b.s == 1) {
            // 1x1 channel projection, accumulated onto the residual output
            conv(cur, wsk[i], nxt, b.ci, b.co, 1, 1, 0, b.szi, b.szo, 1);
        } else {
            // gaussian blur stride-2 (ci == co here), then elementwise add
            const int N = b.co * Po;
            gauss_blur_kernel<<<(N + 255) / 256, 256, 0, stream>>>(
                cur, bufS, b.ci, b.szi, b.szi, b.szi, b.szo, b.szo, b.szo);
            add_kernel<<<(N + 255) / 256, 256, 0, stream>>>(nxt, bufS, N);
        }

        float* t = cur; cur = nxt; nxt = t;
    }

    // ---- final conv: k3 p0, 1280 -> 288 ch, 4^3 -> 2^3 --------------------
    conv(cur, wf, bufR, 1280, 288, 3, 1, 0, 4, 2, 0);

    // ---- quotient-Fourier pooling -> [32, 2, 2, 2] ------------------------
    quotient_pool_kernel<<<1, 256, 0, stream>>>(bufR, Aq, Aout, (float*)d_out, 8);
}
